// R_Attn_plus_4964982194750
// MI455X (gfx1250) — matrix operand order repaired and checked by reading, not compiled, not run on hardware
//
#include <hip/hip_runtime.h>
#include <hip/hip_bf16.h>

#define D_DIM 64
#define LROWS 262144

typedef __attribute__((ext_vector_type(16))) __bf16   v16bf;
typedef __attribute__((ext_vector_type(8)))  float    v8f;
typedef __attribute__((ext_vector_type(4)))  unsigned v4u;
typedef __attribute__((ext_vector_type(8)))  unsigned v8u;

// ---------------- workspace layout (byte offsets) ----------------
#define WS_C1      0         // 64 f32 : W1[:, :64] @ x + W1_b
#define WS_CR      256       // 64 f32 : Wr[:, :64] @ x + Wr_b
#define WS_U       512       // 64 f32 : U @ x
#define WS_MS      768       // 2 f32  : softmax M, 1/S
#define WS_W1P     1024      // 4096 bf16 : W1t packed for WMMA B operand
#define WS_WRP     9216      // 4096 bf16 : Wrt packed
#define WS_WWP     17408     // 4096 bf16 : W  packed
#define WS_BMAX    25600     // 128 f32 : per-block score max
#define WS_SEGA    26112     // 256 f32 : per-block scan segment A
#define WS_SEGB    27136     // 256*64 f32 : per-block scan segment B
#define WS_SCORES  92672     // L f32 raw scores
// total = 92672 + 4*262144 = 1,141,248 bytes

#define K1_TILES 16
#define K1_ROWS_PER_BLOCK (8 * 16 * K1_TILES)        // 2048
#define NB1 (LROWS / K1_ROWS_PER_BLOCK)              // 128

#define K3_TILES 8
#define K3_ROWS_PER_WAVE (16 * K3_TILES)             // 128 contiguous rows per wave
#define K3_ROWS_PER_BLOCK (8 * K3_ROWS_PER_WAVE)     // 1024
#define NB3 (LROWS / K3_ROWS_PER_BLOCK)              // 256

#define LOG2E 1.4426950408889634f

// ---- branch-free fast transcendentals (hardware TRANS ops) ----
__device__ __forceinline__ float fexp2(float x) {
#if __has_builtin(__builtin_amdgcn_exp2f)
  return __builtin_amdgcn_exp2f(x);
#else
  return exp2f(x);
#endif
}
__device__ __forceinline__ float frcp(float x) {
#if __has_builtin(__builtin_amdgcn_rcpf)
  return __builtin_amdgcn_rcpf(x);
#else
  return 1.f / x;
#endif
}
__device__ __forceinline__ float fast_tanh(float x) {
#if __has_builtin(__builtin_amdgcn_tanhf)
  return __builtin_amdgcn_tanhf(x);   // V_TANH_F32
#else
  const float xc = fminf(fmaxf(x, -9.f), 9.f);
  const float e = fexp2(2.f * LOG2E * xc);
  return (e - 1.f) * frcp(e + 1.f);
#endif
}
__device__ __forceinline__ float fast_sigmoid(float x) {
  return frcp(1.f + fexp2(-LOG2E * x));
}
__device__ __forceinline__ float fast_exp(float x) { return fexp2(LOG2E * x); }

// Low 32 bits of a generic LDS pointer are the LDS byte offset
// (ISA 10.2 aperture).
__device__ __forceinline__ unsigned lds_off(const void* p) {
  return (unsigned)(unsigned long long)p;
}

// TDM: 2-D tile load, rows x 64 f32, row-major, into LDS.
// D# group0/group1 per CDNA5 ISA.
__device__ __forceinline__ void tdm_load_rows(const float* gsrc, unsigned lds_byte_off,
                                              unsigned rows) {
  unsigned long long ga = (unsigned long long)gsrc;
  unsigned ga_lo = __builtin_amdgcn_readfirstlane((unsigned)ga);
  unsigned ga_hi = __builtin_amdgcn_readfirstlane((unsigned)(ga >> 32));
  unsigned lo    = __builtin_amdgcn_readfirstlane(lds_byte_off);
  v4u g0; v8u g1;
  g0[0] = 1u;                                   // count=1 valid user descriptor
  g0[1] = lo;                                   // lds_addr
  g0[2] = ga_lo;                                // global_addr[31:0]
  g0[3] = (ga_hi & 0x01FFFFFFu) | (2u << 30);   // global_addr[56:32] | type=2
  g1[0] = 2u << 16;                             // data_size = 4 bytes
  g1[1] = 64u << 16;                            // tensor_dim0 = 64
  g1[2] = (rows & 0xFFFFu) << 16;               // tensor_dim1 lo16 (dim0 hi16 = 0)
  g1[3] = (rows >> 16) | (64u << 16);           // tensor_dim1 hi16 | tile_dim0 = 64
  g1[4] = rows & 0xFFFFu;                       // tile_dim1 = rows, tile_dim2 = 0
  g1[5] = 64u;                                  // tensor_dim0_stride = 64 (lo32)
  g1[6] = 0u;
  g1[7] = 0u;
  asm volatile("tensor_load_to_lds %0, %1" :: "s"(g0), "s"(g1) : "memory");
}

__device__ __forceinline__ v8f wmma_bf16(v16bf a, v16bf b, v8f c) {
  return __builtin_amdgcn_wmma_f32_16x16x32_bf16(false, a, false, b, (short)0, c,
                                                 false, false);
}

// Build 16x32 bf16 A operand from a row-major 16x64 f32 LDS tile.
// 16-bit A layout: lane<16 -> M=lane, K in [0..7]+[16..23] of the 32-chunk;
// lane>=16 -> M=lane-16, K in [8..15]+[24..31].
__device__ __forceinline__ v16bf make_a(const float* tile, int lane, int kc) {
  const int m = lane & 15;
  const int half = lane >> 4;
  const float* rp = tile + m * 64 + kc * 32 + half * 8;
  v16bf a;
#pragma unroll
  for (int e = 0; e < 8; e++) a[e] = (__bf16)rp[e];
#pragma unroll
  for (int e = 0; e < 8; e++) a[8 + e] = (__bf16)rp[16 + e];
  return a;
}

// -------- kernel 0: fold x into constants, pack weight B operands --------
__global__ void __launch_bounds__(256)
prep_kernel(const float* __restrict__ x,
            const float* __restrict__ W1w, const float* __restrict__ W1b,
            const float* __restrict__ Wrw, const float* __restrict__ Wrb,
            const float* __restrict__ Uw,  const float* __restrict__ Ww,
            unsigned char* __restrict__ ws) {
  const int tid = threadIdx.x;
  float* c1 = (float*)(ws + WS_C1);
  float* cr = (float*)(ws + WS_CR);
  float* u  = (float*)(ws + WS_U);
  if (tid < D_DIM) {
    float a = 0.f, b = 0.f, c = 0.f;
    for (int k = 0; k < D_DIM; k++) {
      const float xv = x[k];
      a += W1w[tid * 128 + k] * xv;
      b += Wrw[tid * 128 + k] * xv;
      c += Uw[tid * 64 + k] * xv;
    }
    c1[tid] = a + W1b[tid];
    cr[tid] = b + Wrb[tid];
    u[tid]  = c;
  }
  // B-operand packing: lanes 0-15 hold K 0..7,16..23, lanes 16-31 hold K 8..15,24..31;
  // within a lane element e <-> K = 16*(e>>3) + 8*(lane>>4) + (e&7); N = ntile*16 + (lane&15).
  __bf16* w1p = (__bf16*)(ws + WS_W1P);
  __bf16* wrp = (__bf16*)(ws + WS_WRP);
  __bf16* wwp = (__bf16*)(ws + WS_WWP);
  for (int j = tid; j < 4096; j += blockDim.x) {
    const int e    = j & 15;
    const int lane = (j >> 4) & 31;
    const int kc   = (j >> 9) & 1;
    const int nt   = j >> 10;
    const int n = nt * 16 + (lane & 15);
    const int k = kc * 32 + ((e & 8) << 1) + ((lane & 16) >> 1) + (e & 7);
    w1p[j] = (__bf16)W1w[n * 128 + 64 + k];
    wrp[j] = (__bf16)Wrw[n * 128 + 64 + k];
    wwp[j] = (__bf16)Ww[n * 64 + k];
  }
}

// -------- kernel 1: raw scores via WMMA, per-block max --------
__global__ void __launch_bounds__(256)
score_kernel(const float* __restrict__ tgt, const float* __restrict__ W2w,
             const float* __restrict__ W2b, unsigned char* __restrict__ ws) {
  __shared__ float tile[2][8][16 * 64];    // ping-pong per wave
  __shared__ float red[256];
  const int tid = threadIdx.x, lane = tid & 31, wave = tid >> 5;
  const float* c1 = (const float*)(ws + WS_C1);
  float* scores   = (float*)(ws + WS_SCORES);
  float* bmax     = (float*)(ws + WS_BMAX);
  const v16bf* w1p = (const v16bf*)(ws + WS_W1P);

  v16bf b[4][2];
#pragma unroll
  for (int nt = 0; nt < 4; nt++)
#pragma unroll
    for (int kc = 0; kc < 2; kc++) b[nt][kc] = w1p[(nt * 2 + kc) * 32 + lane];

  const int nidx = lane & 15;
  float c1v[4], w2v[4];
#pragma unroll
  for (int nt = 0; nt < 4; nt++) {
    c1v[nt] = c1[nt * 16 + nidx];
    w2v[nt] = W2w[nt * 16 + nidx];
  }
  const float w2b = W2b[0];

  const int waveRow0 = blockIdx.x * K1_ROWS_PER_BLOCK + wave * (16 * K1_TILES);
  float* buf0 = &tile[0][wave][0];
  float* buf1 = &tile[1][wave][0];
  const unsigned off0 = lds_off(buf0), off1 = lds_off(buf1);

  // preload tile 0
  tdm_load_rows(tgt + (size_t)waveRow0 * 64, off0, 16u);

  float wmax = -3.4e38f;
  for (int t = 0; t < K1_TILES; t++) {
    const int row0 = waveRow0 + t * 16;
    if (t + 1 < K1_TILES) {   // issue next DMA, wait for current only
      tdm_load_rows(tgt + (size_t)(row0 + 16) * 64, (t & 1) ? off0 : off1, 16u);
      __builtin_amdgcn_s_wait_tensorcnt(1);
    } else {
      __builtin_amdgcn_s_wait_tensorcnt(0);
    }
    asm volatile("" ::: "memory");
    const float* cur = (t & 1) ? buf1 : buf0;

    const v16bf a0 = make_a(cur, lane, 0);
    const v16bf a1 = make_a(cur, lane, 1);
    float sacc[8] = {0.f, 0.f, 0.f, 0.f, 0.f, 0.f, 0.f, 0.f};
#pragma unroll
    for (int nt = 0; nt < 4; nt++) {
      v8f c = {0.f, 0.f, 0.f, 0.f, 0.f, 0.f, 0.f, 0.f};
      c = wmma_bf16(a0, b[nt][0], c);
      c = wmma_bf16(a1, b[nt][1], c);
#pragma unroll
      for (int r = 0; r < 8; r++) sacc[r] += fast_tanh(c[r] + c1v[nt]) * w2v[nt];
    }
    // reduce over the 16 lanes of each half (N dimension)
#pragma unroll
    for (int r = 0; r < 8; r++) {
      sacc[r] += __shfl_xor(sacc[r], 1, 32);
      sacc[r] += __shfl_xor(sacc[r], 2, 32);
      sacc[r] += __shfl_xor(sacc[r], 4, 32);
      sacc[r] += __shfl_xor(sacc[r], 8, 32);
      sacc[r] += w2b;
    }
    if ((lane & 15) == 0) {        // lanes 0 (rows 0..7) and 16 (rows 8..15)
      const int mb = (lane >> 4) * 8;
#pragma unroll
      for (int r = 0; r < 8; r++) {
        scores[row0 + mb + r] = sacc[r];
        wmax = fmaxf(wmax, sacc[r]);
      }
    }
  }
  red[tid] = wmax;
  __syncthreads();
  for (int s = 128; s > 0; s >>= 1) {
    if (tid < s) red[tid] = fmaxf(red[tid], red[tid + s]);
    __syncthreads();
  }
  if (tid == 0) bmax[blockIdx.x] = red[0];
}

// -------- kernel 2: softmax global max + sum --------
__global__ void __launch_bounds__(1024)
softmax_reduce_kernel(unsigned char* __restrict__ ws) {
  __shared__ float red[1024];
  const float* scores = (const float*)(ws + WS_SCORES);
  const float* bmax   = (const float*)(ws + WS_BMAX);
  float* ms = (float*)(ws + WS_MS);
  const int tid = threadIdx.x;
  float m = -3.4e38f;
  for (int i = tid; i < NB1; i += 1024) m = fmaxf(m, bmax[i]);
  red[tid] = m;
  __syncthreads();
  for (int s = 512; s > 0; s >>= 1) {
    if (tid < s) red[tid] = fmaxf(red[tid], red[tid + s]);
    __syncthreads();
  }
  const float M = red[0];
  __syncthreads();
  float acc = 0.f;
  for (int i = tid; i < LROWS; i += 1024) acc += fast_exp(scores[i] - M);
  red[tid] = acc;
  __syncthreads();
  for (int s = 512; s > 0; s >>= 1) {
    if (tid < s) red[tid] += red[tid + s];
    __syncthreads();
  }
  if (tid == 0) { ms[0] = M; ms[1] = 1.f / red[0]; }
}

// -------- kernel 3: r, l, attn, ordered per-block affine-scan segment --------
__global__ void __launch_bounds__(256)
fuse_kernel(const float* __restrict__ tgt, const float* __restrict__ Wb,
            float* __restrict__ out, unsigned char* __restrict__ ws) {
  __shared__ float tile[2][8][16 * 64];    // ping-pong per wave
  __shared__ float lmat[8][16 * 64];
  __shared__ float pbuf[8][16];
  __shared__ float segA[8];
  __shared__ float segB[8][64];
  const int tid = threadIdx.x, lane = tid & 31, wave = tid >> 5;
  const float* cr = (const float*)(ws + WS_CR);
  const float* u  = (const float*)(ws + WS_U);
  const float* ms = (const float*)(ws + WS_MS);
  const float* scores = (const float*)(ws + WS_SCORES);
  const v16bf* wrp = (const v16bf*)(ws + WS_WRP);
  const v16bf* wwp = (const v16bf*)(ws + WS_WWP);
  float* gsegA = (float*)(ws + WS_SEGA);
  float* gsegB = (float*)(ws + WS_SEGB);

  v16bf br[4][2], bw[4][2];
#pragma unroll
  for (int nt = 0; nt < 4; nt++)
#pragma unroll
    for (int kc = 0; kc < 2; kc++) {
      br[nt][kc] = wrp[(nt * 2 + kc) * 32 + lane];
      bw[nt][kc] = wwp[(nt * 2 + kc) * 32 + lane];
    }
  const int nidx = lane & 15;
  float crv[4], uv[4], wbv[4];
#pragma unroll
  for (int nt = 0; nt < 4; nt++) {
    crv[nt] = cr[nt * 16 + nidx];
    uv[nt]  = u[nt * 16 + nidx];
    wbv[nt] = Wb[nt * 16 + nidx];
  }
  const float M = ms[0], invS = ms[1];
  const int half8 = (lane >> 4) * 8;

  const int waveRow0 = blockIdx.x * K3_ROWS_PER_BLOCK + wave * K3_ROWS_PER_WAVE;
  float* buf0 = &tile[0][wave][0];
  float* buf1 = &tile[1][wave][0];
  float* myl  = &lmat[wave][0];
  float* myp  = &pbuf[wave][0];
  const unsigned off0 = lds_off(buf0), off1 = lds_off(buf1);
  const int c0 = lane * 2, c1i = lane * 2 + 1;

  // preload tile 0
  tdm_load_rows(tgt + (size_t)waveRow0 * 64, off0, 16u);

  float Arun = 1.f, B0 = 0.f, B1 = 0.f;
  for (int t = 0; t < K3_TILES; t++) {
    const int row0 = waveRow0 + t * 16;
    if (t + 1 < K3_TILES) {   // issue next DMA, wait for current only
      tdm_load_rows(tgt + (size_t)(row0 + 16) * 64, (t & 1) ? off0 : off1, 16u);
      __builtin_amdgcn_s_wait_tensorcnt(1);
    } else {
      __builtin_amdgcn_s_wait_tensorcnt(0);
    }
    asm volatile("" ::: "memory");
    const float* curt = (t & 1) ? buf1 : buf0;

    const v16bf a0 = make_a(curt, lane, 0);
    const v16bf a1 = make_a(curt, lane, 1);
#pragma unroll
    for (int nt = 0; nt < 4; nt++) {
      v8f cR = {0.f, 0.f, 0.f, 0.f, 0.f, 0.f, 0.f, 0.f};
      cR = wmma_bf16(a0, br[nt][0], cR);
      cR = wmma_bf16(a1, br[nt][1], cR);
      v8f cW = {0.f, 0.f, 0.f, 0.f, 0.f, 0.f, 0.f, 0.f};
      cW = wmma_bf16(a0, bw[nt][0], cW);
      cW = wmma_bf16(a1, bw[nt][1], cW);
      const int n = nt * 16 + nidx;
#pragma unroll
      for (int r = 0; r < 8; r++) {
        const float rv = fast_sigmoid(cR[r] + crv[nt]);
        const float lv = fast_tanh(cW[r] + wbv[nt] + rv * uv[nt]);
        myl[(half8 + r) * 64 + n] = lv;
      }
    }
    if (lane < 16) {
      const float pv = fast_exp(scores[row0 + lane] - M) * invS;
      myp[lane] = pv;
      out[64 + row0 + lane] = pv;   // attn_weights output
    }
    // ordered affine scan over the 16 rows of this tile (same-wave LDS,
    // lockstep; compiler inserts dscnt waits for the RAW dependences)
#pragma unroll
    for (int m = 0; m < 16; m++) {
      const float p = myp[m], q = 1.f - p;
      B0 = q * B0 + p * myl[m * 64 + c0];
      B1 = q * B1 + p * myl[m * 64 + c1i];
      Arun *= q;
    }
  }
  segB[wave][c0]  = B0;
  segB[wave][c1i] = B1;
  if (lane == 0) segA[wave] = Arun;
  __syncthreads();
  if (tid < 64) {   // fold the 8 ordered wave segments -> block segment
    float a = 1.f, bb = 0.f;
#pragma unroll
    for (int w = 0; w < 8; w++) {
      bb = segA[w] * bb + segB[w][tid];
      a *= segA[w];
    }
    gsegB[blockIdx.x * 64 + tid] = bb;
    if (tid == 0) gsegA[blockIdx.x] = a;
  }
}

// -------- kernel 4: fold block segments -> hidden --------
__global__ void __launch_bounds__(64)
final_kernel(float* __restrict__ out, const unsigned char* __restrict__ ws) {
  const float* gsegA = (const float*)(ws + WS_SEGA);
  const float* gsegB = (const float*)(ws + WS_SEGB);
  const int c = threadIdx.x;
  if (c < 64) {
    float h = 0.f;
    for (int i = 0; i < NB3; i++) h = gsegA[i] * h + gsegB[i * 64 + c];
    out[c] = h;
  }
}

extern "C" void kernel_launch(void* const* d_in, const int* in_sizes, int n_in,
                              void* d_out, int out_size, void* d_ws, size_t ws_size,
                              hipStream_t stream) {
  (void)in_sizes; (void)n_in; (void)out_size; (void)ws_size;
  const float* x   = (const float*)d_in[0];
  const float* tgt = (const float*)d_in[1];
  const float* W1w = (const float*)d_in[2];
  const float* W1b = (const float*)d_in[3];
  const float* Wrw = (const float*)d_in[4];
  const float* Wrb = (const float*)d_in[5];
  const float* Uw  = (const float*)d_in[6];
  const float* W2w = (const float*)d_in[7];
  const float* W2b = (const float*)d_in[8];
  const float* Ww  = (const float*)d_in[9];
  const float* Wb  = (const float*)d_in[10];
  float* out = (float*)d_out;
  unsigned char* ws = (unsigned char*)d_ws;

  hipLaunchKernelGGL(prep_kernel, dim3(1), dim3(256), 0, stream,
                     x, W1w, W1b, Wrw, Wrb, Uw, Ww, ws);
  hipLaunchKernelGGL(score_kernel, dim3(NB1), dim3(256), 0, stream,
                     tgt, W2w, W2b, ws);
  hipLaunchKernelGGL(softmax_reduce_kernel, dim3(1), dim3(1024), 0, stream, ws);
  hipLaunchKernelGGL(fuse_kernel, dim3(NB3), dim3(256), 0, stream,
                     tgt, Wb, out, ws);
  hipLaunchKernelGGL(final_kernel, dim3(1), dim3(64), 0, stream, out, ws);
}
